// FMoE_59742995087815
// MI455X (gfx1250) — matrix operand order repaired and checked by reading, not compiled, not run on hardware
//
#include <hip/hip_runtime.h>
#include <hip/hip_bf16.h>

// ---------------------------------------------------------------------------
// Problem constants (match the reference)
// ---------------------------------------------------------------------------
constexpr int Ntok = 16384;
constexpr int Dm   = 1024;   // model dim
constexpr int Ex   = 16;     // experts
constexpr int Kt   = 2;      // top-k
constexpr int Hd   = 1024;   // hidden dim
constexpr int CAP  = 4096;   // expert capacity

// WMMA types (CDNA5, wave32)
typedef __attribute__((ext_vector_type(16))) __bf16 v16bf;
typedef __attribute__((ext_vector_type(8)))  float  v8f;
typedef __attribute__((ext_vector_type(4)))  int    v4i;

// Async global->LDS path (gfx1250); falls back to register staging if absent.
#if defined(__AMDGCN__) && \
    __has_builtin(__builtin_amdgcn_global_load_async_to_lds_b128) && \
    __has_builtin(__builtin_amdgcn_s_wait_asynccnt)
#define USE_ASYNC_LDS 1
#else
#define USE_ASYNC_LDS 0
#endif

#if USE_ASYNC_LDS
#define GAS1(p) ((__attribute__((address_space(1))) v4i*)(p))
#define LAS3(p) ((__attribute__((address_space(3))) v4i*)(p))
#endif

// ---------------------------------------------------------------------------
// fp32 -> bf16 bits (round to nearest even)
// ---------------------------------------------------------------------------
__device__ inline unsigned short f2bf(float f) {
    unsigned int u = __float_as_uint(f);
    unsigned int lsb = (u >> 16) & 1u;
    u += 0x7fffu + lsb;
    return (unsigned short)(u >> 16);
}

__device__ inline v16bf load_frag(const unsigned short* p0, const unsigned short* p1) {
    union { uint4 q[2]; v16bf v; } u;
    u.q[0] = *(const uint4*)p0;
    u.q[1] = *(const uint4*)p1;
    return u.v;
}

// ---------------------------------------------------------------------------
// init: zero out, zero counters, slot2row = -1
// ---------------------------------------------------------------------------
__global__ __launch_bounds__(256) void init_kernel(float* out, int* slot2row, int* cnt) {
    size_t i = (size_t)blockIdx.x * 256 + threadIdx.x;
    if (i < (size_t)Ntok * Dm) out[i] = 0.0f;
    if (i < (size_t)Ex * CAP)  slot2row[i] = -1;
    if (i < Ex)                cnt[i] = 0;
}

// ---------------------------------------------------------------------------
// Transposing fp32 -> bf16 weight conversion.
// src: [E][R][C] fp32 ; dst: [E][C][R] bf16   (tiled 32x32 via LDS)
// ---------------------------------------------------------------------------
__global__ __launch_bounds__(256) void cvtT_kernel(const float* __restrict__ src,
                                                   unsigned short* __restrict__ dst,
                                                   int R, int C) {
    __shared__ float tile[32][33];
    const int e  = blockIdx.z;
    const int c0 = blockIdx.x * 32;
    const int r0 = blockIdx.y * 32;
    const int tx = threadIdx.x & 31;
    const int ty = threadIdx.x >> 5;               // 0..7
    const float* s = src + (size_t)e * R * C;
    unsigned short* d = dst + (size_t)e * R * C;
#pragma unroll
    for (int i = 0; i < 4; ++i)
        tile[ty + 8 * i][tx] = s[(size_t)(r0 + ty + 8 * i) * C + c0 + tx];
    __syncthreads();
#pragma unroll
    for (int i = 0; i < 4; ++i)
        d[(size_t)(c0 + ty + 8 * i) * R + r0 + tx] = f2bf(tile[tx][ty + 8 * i]);
}

// ---------------------------------------------------------------------------
// Gating: one wave (32 lanes) per token. logits = x@Wg + bg ; top-2 ; softmax2
// ---------------------------------------------------------------------------
__global__ __launch_bounds__(256) void gate_kernel(const float* __restrict__ x,
                                                   const float* __restrict__ Wg,
                                                   const float* __restrict__ bg,
                                                   int*   __restrict__ e_sel,
                                                   float* __restrict__ gate) {
    const int wid  = threadIdx.x >> 5;
    const int lane = threadIdx.x & 31;
    const int n    = blockIdx.x * 8 + wid;
    if (n >= Ntok) return;

    float acc[Ex];
#pragma unroll
    for (int e = 0; e < Ex; ++e) acc[e] = 0.0f;

    const float* xr = x + (size_t)n * Dm;
    for (int d = lane; d < Dm; d += 32) {
        float xv = xr[d];
        const float* wr = Wg + (size_t)d * Ex;
#pragma unroll
        for (int e = 0; e < Ex; ++e) acc[e] += xv * wr[e];
    }
#pragma unroll
    for (int e = 0; e < Ex; ++e) {
#pragma unroll
        for (int off = 16; off > 0; off >>= 1)
            acc[e] += __shfl_xor(acc[e], off, 32);
    }
    if (lane == 0) {
#pragma unroll
        for (int e = 0; e < Ex; ++e) acc[e] += bg[e];
        float v0 = -3.4e38f; int i0 = 0;
#pragma unroll
        for (int e = 0; e < Ex; ++e) if (acc[e] > v0) { v0 = acc[e]; i0 = e; }
        float v1 = -3.4e38f; int i1 = 0;
#pragma unroll
        for (int e = 0; e < Ex; ++e) if (e != i0 && acc[e] > v1) { v1 = acc[e]; i1 = e; }
        float g0 = 1.0f / (1.0f + __expf(v1 - v0));
        float g1 = 1.0f - g0;
        e_sel[n * 2 + 0] = i0;  gate[n * 2 + 0] = g0;
        e_sel[n * 2 + 1] = i1;  gate[n * 2 + 1] = g1;
    }
}

// ---------------------------------------------------------------------------
// Scatter: one wave per (token, k). Atomic slot grab + bf16 row copy.
// ---------------------------------------------------------------------------
__global__ __launch_bounds__(256) void scatter_kernel(const float* __restrict__ x,
                                                      const int*   __restrict__ e_sel,
                                                      int*  __restrict__ cnt,
                                                      int*  __restrict__ slot2row,
                                                      unsigned short* __restrict__ xbuf) {
    const int wid  = threadIdx.x >> 5;
    const int lane = threadIdx.x & 31;
    const int idx  = blockIdx.x * 8 + wid;          // 0 .. N*K-1
    if (idx >= Ntok * Kt) return;
    const int n = idx >> 1;
    const int e = e_sel[idx];

    int slot = 0;
    if (lane == 0) slot = atomicAdd(&cnt[e], 1);
    slot = __shfl(slot, 0, 32);
    if (slot >= CAP) return;                        // drop (prob ~0)
    if (lane == 0) slot2row[e * CAP + slot] = idx;

    const float* src = x + (size_t)n * Dm;
    unsigned short* dst = xbuf + ((size_t)e * CAP + slot) * Dm;
    for (int c = lane * 4; c < Dm; c += 128) {
        float4 v = *(const float4*)(src + c);
        ushort4 o;
        o.x = f2bf(v.x); o.y = f2bf(v.y); o.z = f2bf(v.z); o.w = f2bf(v.w);
        *(ushort4*)(dst + c) = o;
    }
}

// ---------------------------------------------------------------------------
// bf16 WMMA grouped GEMM, 128x128 tile / 256 threads (8 waves, wave32).
// Wave grid 4(M) x 2(N); each wave: 32x64 = 2x4 C tiles of 16x16.
// B is PRE-TRANSPOSED ([N][K] bf16) so both panels stage as b128 row copies.
// Async double-buffered global->LDS staging when available.
// MODE 1: out = bf16( gelu(acc + b1) )       -> hidden
// MODE 2: atomicAdd( out[token], gate * (acc + b2) )  (gather+combine fused)
// ---------------------------------------------------------------------------
constexpr int KD  = 1024;    // reduction dim
constexpr int NC  = 1024;    // output cols
constexpr int KS  = KD / 32; // K steps
constexpr int LDT = 40;      // padded LDS row stride (bf16 elems), 80B

template <int MODE>
__global__ __launch_bounds__(256) void gemm_kernel(const unsigned short* __restrict__ A,
                                                   const unsigned short* __restrict__ Bt,
                                                   const float* __restrict__ bias,
                                                   unsigned short* __restrict__ Hout,
                                                   float* __restrict__ Out,
                                                   const int* __restrict__ cnt,
                                                   const int* __restrict__ slot2row,
                                                   const float* __restrict__ gate) {
    __shared__ unsigned short As[2][128 * LDT];
    __shared__ unsigned short Bs[2][128 * LDT];

    const int e  = blockIdx.z;
    const int m0 = blockIdx.y * 128;
    const int n0 = blockIdx.x * 128;

    int mcnt = cnt[e]; if (mcnt > CAP) mcnt = CAP;
    if (m0 >= mcnt) return;                         // skip empty tiles

    const unsigned short* Ae  = A  + ((size_t)e * CAP + m0) * KD;
    const unsigned short* Bte = Bt + ((size_t)e * NC + n0) * KD;   // [N][K] rows

    const int tid  = threadIdx.x;
    const int lane = tid & 31;
    const int wid  = tid >> 5;
    const int wm   = wid >> 1;                      // 0..3
    const int wn   = wid & 1;                       // 0..1
    const int half = lane >> 4;
    const int ln   = lane & 15;

    // stage-load indices: each thread copies one 32B half-row for A and B
    const int ar = tid >> 1;                        // row 0..127
    const int ah = tid & 1;                         // 16-elem half

    v8f c[2][4];
#pragma unroll
    for (int mi = 0; mi < 2; ++mi)
#pragma unroll
        for (int ni = 0; ni < 4; ++ni)
#pragma unroll
            for (int j = 0; j < 8; ++j) c[mi][ni][j] = 0.0f;

#if USE_ASYNC_LDS
    auto stage_async = [&](int kk, int b) {
        const unsigned short* sA = Ae + (size_t)ar * KD + kk * 32 + ah * 16;
        unsigned short* dA = &As[b][ar * LDT + ah * 16];
        __builtin_amdgcn_global_load_async_to_lds_b128(GAS1(sA),     LAS3(dA),     0, 0);
        __builtin_amdgcn_global_load_async_to_lds_b128(GAS1(sA + 8), LAS3(dA + 8), 0, 0);
        const unsigned short* sB = Bte + (size_t)ar * KD + kk * 32 + ah * 16;
        unsigned short* dB = &Bs[b][ar * LDT + ah * 16];
        __builtin_amdgcn_global_load_async_to_lds_b128(GAS1(sB),     LAS3(dB),     0, 0);
        __builtin_amdgcn_global_load_async_to_lds_b128(GAS1(sB + 8), LAS3(dB + 8), 0, 0);
    };
    stage_async(0, 0);
#endif

    for (int kk = 0; kk < KS; ++kk) {
#if USE_ASYNC_LDS
        const int b = kk & 1;
        __builtin_amdgcn_s_wait_asynccnt(0);        // own tile-kk copies done
        __syncthreads();                            // everyone's copies + prev compute done
        if (kk + 1 < KS) stage_async(kk + 1, b ^ 1);
#else
        const int b = 0;
        __syncthreads();
        {
            const unsigned short* sA = Ae + (size_t)ar * KD + kk * 32 + ah * 16;
            unsigned short* dA = &As[0][ar * LDT + ah * 16];
            ((uint4*)dA)[0] = ((const uint4*)sA)[0];
            ((uint4*)dA)[1] = ((const uint4*)sA)[1];
            const unsigned short* sB = Bte + (size_t)ar * KD + kk * 32 + ah * 16;
            unsigned short* dB = &Bs[0][ar * LDT + ah * 16];
            ((uint4*)dB)[0] = ((const uint4*)sB)[0];
            ((uint4*)dB)[1] = ((const uint4*)sB)[1];
            if (kk + 1 < KS) {
                __builtin_prefetch(sA + 32, 0, 0);
                __builtin_prefetch(sB + 32, 0, 0);
            }
        }
        __syncthreads();
#endif
        // --- fragments ---
        v16bf af[2], bfr[4];
#pragma unroll
        for (int mi = 0; mi < 2; ++mi) {
            const unsigned short* p = &As[b][(wm * 32 + mi * 16 + ln) * LDT];
            // A lane layout: K = half*8 .. +7  and  16+half*8 .. +7
            af[mi] = load_frag(p + half * 8, p + 16 + half * 8);
        }
#pragma unroll
        for (int ni = 0; ni < 4; ++ni) {
            const unsigned short* p = &Bs[b][(wn * 64 + ni * 16 + ln) * LDT + half * 8];
            // B lane layout: K = half*8 .. +7  and  16+half*8 .. +7
            bfr[ni] = load_frag(p, p + 16);
        }
#pragma unroll
        for (int mi = 0; mi < 2; ++mi)
#pragma unroll
            for (int ni = 0; ni < 4; ++ni)
                c[mi][ni] = __builtin_amdgcn_wmma_f32_16x16x32_bf16(
                    false, af[mi], false, bfr[ni], (short)0, c[mi][ni], false, false);
    }

    // --- epilogue ---
#pragma unroll
    for (int mi = 0; mi < 2; ++mi) {
#pragma unroll
        for (int ni = 0; ni < 4; ++ni) {
#pragma unroll
            for (int j = 0; j < 8; ++j) {
                const int rl   = wm * 32 + mi * 16 + half * 8 + j;
                const int cl   = wn * 64 + ni * 16 + ln;
                const int grow = m0 + rl;
                const int gcol = n0 + cl;
                float val = c[mi][ni][j];
                if (MODE == 1) {
                    if (grow < mcnt) {
                        val += bias[e * NC + gcol];
                        // tanh-approx GELU (jax.nn.gelu default)
                        float u = 0.7978845608028654f * (val + 0.044715f * val * val * val);
                        float g = 0.5f * val * (1.0f + tanhf(u));
                        Hout[((size_t)e * CAP + grow) * NC + gcol] = f2bf(g);
                    }
                } else {
                    const int idx = slot2row[e * CAP + grow];
                    if (idx >= 0) {
                        val += bias[e * NC + gcol];
                        atomicAdd(&Out[(size_t)(idx >> 1) * NC + gcol], gate[idx] * val);
                    }
                }
            }
        }
    }
}

// ---------------------------------------------------------------------------
// Host-side launch
// ---------------------------------------------------------------------------
extern "C" void kernel_launch(void* const* d_in, const int* in_sizes, int n_in,
                              void* d_out, int out_size, void* d_ws, size_t ws_size,
                              hipStream_t stream) {
    const float* x  = (const float*)d_in[0];   // [N, D]
    const float* Wg = (const float*)d_in[1];   // [D, E]
    const float* bg = (const float*)d_in[2];   // [E]
    const float* W1 = (const float*)d_in[3];   // [E, D, H]
    const float* b1 = (const float*)d_in[4];   // [E, H]
    const float* W2 = (const float*)d_in[5];   // [E, H, D]
    const float* b2 = (const float*)d_in[6];   // [E, D]
    float* out = (float*)d_out;                // [N, D]

    // workspace carve-out (256B aligned sections)
    char* ws = (char*)d_ws;
    size_t off = 0;
    auto take = [&](size_t bytes) -> char* {
        char* p = ws + off;
        off = (off + bytes + 255) & ~(size_t)255;
        return p;
    };
    int*   cnt      = (int*)  take(Ex * sizeof(int));
    int*   e_sel    = (int*)  take((size_t)Ntok * Kt * sizeof(int));
    float* gate     = (float*)take((size_t)Ntok * Kt * sizeof(float));
    int*   slot2row = (int*)  take((size_t)Ex * CAP * sizeof(int));
    unsigned short* w1t  = (unsigned short*)take((size_t)Ex * Dm * Hd * 2);  // [E][H][D]
    unsigned short* w2t  = (unsigned short*)take((size_t)Ex * Hd * Dm * 2);  // [E][D][H]
    unsigned short* xbuf = (unsigned short*)take((size_t)Ex * CAP * Dm * 2);
    unsigned short* hbuf = (unsigned short*)take((size_t)Ex * CAP * Hd * 2);
    (void)ws_size; (void)in_sizes; (void)n_in; (void)out_size;

    // 1) init (zero out/counters, slot2row = -1)
    init_kernel<<<(Ntok * (size_t)Dm) / 256, 256, 0, stream>>>(out, slot2row, cnt);

    // 2) weight conversion fp32 -> bf16, transposed to [N][K] for b128 staging
    dim3 t1(Hd / 32, Dm / 32, Ex);                 // W1 [E][D][H] -> w1t [E][H][D]
    cvtT_kernel<<<t1, 256, 0, stream>>>(W1, w1t, Dm, Hd);
    dim3 t2(Dm / 32, Hd / 32, Ex);                 // W2 [E][H][D] -> w2t [E][D][H]
    cvtT_kernel<<<t2, 256, 0, stream>>>(W2, w2t, Hd, Dm);

    // 3) gating (one wave per token)
    gate_kernel<<<Ntok / 8, 256, 0, stream>>>(x, Wg, bg, e_sel, gate);

    // 4) scatter (one wave per token-copy)
    scatter_kernel<<<(Ntok * Kt) / 8, 256, 0, stream>>>(x, e_sel, cnt, slot2row, xbuf);

    // 5) GEMM1: hidden = gelu(xbuf @ W1 + b1)       [bf16 WMMA]
    dim3 g1(Hd / 128, CAP / 128, Ex);
    gemm_kernel<1><<<g1, 256, 0, stream>>>(xbuf, w1t, b1, hbuf, nullptr,
                                           cnt, nullptr, nullptr);

    // 6) GEMM2 + fused gather/combine: out += gate * (hidden @ W2 + b2)
    dim3 g2(Dm / 128, CAP / 128, Ex);
    gemm_kernel<2><<<g2, 256, 0, stream>>>(hbuf, w2t, b2, nullptr, out,
                                           cnt, slot2row, gate);
}
